// Glm4MoeAttention_27333171872042
// MI455X (gfx1250) — matrix operand order repaired and checked by reading, not compiled, not run on hardware
//
#include <hip/hip_runtime.h>
#include <hip/hip_bf16.h>

// ---------- types ----------
typedef __attribute__((ext_vector_type(16))) __bf16 v16bf;
typedef __attribute__((ext_vector_type(8)))  float  v8f;
typedef __attribute__((ext_vector_type(8)))  unsigned int v8u;
typedef __attribute__((ext_vector_type(4)))  int v4i_;
typedef __attribute__((address_space(1))) v4i_* as1_v4i;
typedef __attribute__((address_space(3))) v4i_* as3_v4i;

#define H_   32
#define HKV_ 2
#define D_   128
#define HID_ 4096

#if defined(__has_builtin)
#if __has_builtin(__builtin_amdgcn_global_load_async_to_lds_b128)
#define HAVE_ASYNC_LDS 1
#endif
#endif

__device__ __forceinline__ unsigned short f2bf(float f) {
    unsigned int u = __builtin_bit_cast(unsigned int, f);
    u += 0x7FFFu + ((u >> 16) & 1u);           // round-to-nearest-even
    return (unsigned short)(u >> 16);
}

__device__ __forceinline__ v8f v8f_zero() {
    v8f z;
#pragma unroll
    for (int i = 0; i < 8; ++i) z[i] = 0.0f;
    return z;
}

__device__ __forceinline__ v8f wmma_bf16(v16bf a, v16bf b, v8f c) {
    return __builtin_amdgcn_wmma_f32_16x16x32_bf16(
        false, a, false, b, (short)0, c, false, false);
}

// A fragment (16x32, 16-bit): lane m = lane&15, half = lane>>4.
// VGPR pair j holds K = 16*half + 2*j (+0/+1).
__device__ __forceinline__ v16bf load_a_rowmajor(const unsigned short* rowbase,
                                                 int half) {
    v8u r;
#pragma unroll
    for (int j = 0; j < 8; ++j) {
        int k = 16 * half + 2 * j;
        r[j] = *(const unsigned int*)(rowbase + k);
    }
    return __builtin_bit_cast(v16bf, r);
}

// B fragment (32x16, 16-bit): per lane 16 contiguous K values at K=16*half.
__device__ __forceinline__ v16bf load_b16(const unsigned short* p) {
    v8u r;
#pragma unroll
    for (int j = 0; j < 8; ++j) r[j] = *(const unsigned int*)(p + 2 * j);
    return __builtin_bit_cast(v16bf, r);
}

// 16-byte global -> LDS copy: async (ASYNCcnt-tracked) when available.
__device__ __forceinline__ void cp16_g2l(unsigned short* l,
                                         const unsigned short* g) {
#ifdef HAVE_ASYNC_LDS
    __builtin_amdgcn_global_load_async_to_lds_b128(
        (as1_v4i)(v4i_*)g, (as3_v4i)(v4i_*)l, 0, 0);
#else
    *(int4*)l = *(const int4*)g;
#endif
}

// ---------------------------------------------------------------------------
// fp32 -> bf16 elementwise (packed stores)
// ---------------------------------------------------------------------------
__global__ __launch_bounds__(256) void convert_bf16_kernel(
    const float* __restrict__ src, unsigned short* __restrict__ dst, long n) {
    long i = ((long)blockIdx.x * blockDim.x + threadIdx.x) * 2;
    long stride = (long)gridDim.x * blockDim.x * 2;
    for (; i < n; i += stride) {
        unsigned int pk = (unsigned int)f2bf(src[i]) |
                          ((unsigned int)f2bf(src[i + 1]) << 16);
        *(unsigned int*)(dst + i) = pk;
    }
}

// ---------------------------------------------------------------------------
// W[K][N] fp32 -> Wt[N][K] bf16 (tiled transpose, 32x32 through LDS)
// ---------------------------------------------------------------------------
__global__ __launch_bounds__(256) void transpose_bf16_kernel(
    const float* __restrict__ W, unsigned short* __restrict__ Wt,
    int K, int N) {
    __shared__ float tile[32][33];
    const int tx = threadIdx.x & 31, ty = threadIdx.x >> 5;  // 32 x 8
    const int k0 = blockIdx.y * 32, n0 = blockIdx.x * 32;
#pragma unroll
    for (int i = 0; i < 4; ++i)
        tile[ty + 8 * i][tx] = W[(size_t)(k0 + ty + 8 * i) * N + n0 + tx];
    __syncthreads();
#pragma unroll
    for (int i = 0; i < 4; ++i) {
        int n = n0 + ty + 8 * i;
        Wt[(size_t)n * K + k0 + tx] = f2bf(tile[tx][ty + 8 * i]);
    }
}

// ---------------------------------------------------------------------------
// GEMM: C[M,N] = A[M,K] @ Bt[N,K]^T (+bias). bf16 in, fp32 WMMA accum.
// Block tile 128x256, 8 waves (2 M x 4 N), per-wave 64x64 = 4x4 fragments
// (16 WMMAs per K-step from 4+4 fragment loads). K step 32.
// Double-buffered LDS staged with async global->LDS b128 copies.
// ---------------------------------------------------------------------------
template <bool HAS_BIAS>
__global__ __launch_bounds__(256) void gemm_bf16t_kernel(
    const unsigned short* __restrict__ A, const unsigned short* __restrict__ Bt,
    const float* __restrict__ bias, float* __restrict__ C,
    int M, int N, int K) {
    __shared__ unsigned short sA[2][128 * 32];  // [m][k]  8 KB per buffer
    __shared__ unsigned short sB[2][256 * 32];  // [n][k] 16 KB per buffer

    const int tid = threadIdx.x;
    const int lane = tid & 31, wid = tid >> 5;
    const int wm = wid & 1, wn = wid >> 1;       // 2 x 4 wave grid
    const int bm = blockIdx.y * 128, bn = blockIdx.x * 256;
    const int half = lane >> 4, l16 = lane & 15;

    // Staging: 16B segments. A tile: 512 segs (2/thread); B: 1024 (4/thread).
    auto stage = [&](int buf, int k0) {
#pragma unroll
        for (int s = 0; s < 2; ++s) {
            int seg = tid + s * 256;
            int r = seg >> 2, o = (seg & 3) * 8;
            cp16_g2l(&sA[buf][r * 32 + o], A + (size_t)(bm + r) * K + k0 + o);
        }
#pragma unroll
        for (int s = 0; s < 4; ++s) {
            int seg = tid + s * 256;
            int r = seg >> 2, o = (seg & 3) * 8;
            cp16_g2l(&sB[buf][r * 32 + o], Bt + (size_t)(bn + r) * K + k0 + o);
        }
    };

    v8f acc[4][4];
#pragma unroll
    for (int i = 0; i < 4; ++i)
#pragma unroll
        for (int j = 0; j < 4; ++j) acc[i][j] = v8f_zero();

    int buf = 0;
    stage(0, 0);                                   // ASYNCcnt += 6
    for (int k0 = 0; k0 < K; k0 += 32) {
        const bool more = (k0 + 32) < K;
        if (more) stage(buf ^ 1, k0 + 32);         // up to 12 outstanding
        if (more) asm volatile("s_wait_asynccnt 0x6" ::: "memory");
        else      asm volatile("s_wait_asynccnt 0x0" ::: "memory");
        __syncthreads();

        v16bf afr[4], bfr[4];
#pragma unroll
        for (int i = 0; i < 4; ++i) {
            int m = wm * 64 + i * 16 + l16;
            afr[i] = load_a_rowmajor(&sA[buf][m * 32], half);
        }
#pragma unroll
        for (int j = 0; j < 4; ++j) {
            int n = wn * 64 + j * 16 + l16;
            bfr[j] = load_b16(&sB[buf][n * 32 + 16 * half]);
        }
#pragma unroll
        for (int i = 0; i < 4; ++i)
#pragma unroll
            for (int j = 0; j < 4; ++j)
                acc[i][j] = wmma_bf16(afr[i], bfr[j], acc[i][j]);
        __syncthreads();
        buf ^= 1;
    }

#pragma unroll
    for (int i = 0; i < 4; ++i)
#pragma unroll
        for (int j = 0; j < 4; ++j) {
            int col = bn + wn * 64 + j * 16 + l16;
            float bv = HAS_BIAS ? bias[col] : 0.0f;
#pragma unroll
            for (int r = 0; r < 8; ++r) {
                int row = bm + wm * 64 + i * 16 + r + 8 * half;
                C[(size_t)row * N + col] = acc[i][j][r] + bv;
            }
        }
}

// ---------------------------------------------------------------------------
// RMSNorm + RoPE + bf16 repack. grid = (T, 36).
// Q [H][T][D], K [HKV][T][D] bf16; V stored transposed [HKV][D][T].
// ---------------------------------------------------------------------------
__global__ __launch_bounds__(128) void normrope_kernel(
    const float* __restrict__ qkv, const int* __restrict__ positions,
    const float* __restrict__ qw, const float* __restrict__ kw,
    unsigned short* __restrict__ Q, unsigned short* __restrict__ Km,
    unsigned short* __restrict__ Vt, int T) {
    __shared__ float sh[128];
    __shared__ float red[4];
    const int t = blockIdx.x;
    const int slot = blockIdx.y;
    const int d = threadIdx.x;

    float x = qkv[(size_t)t * 4608 + slot * 128 + d];

    if (slot >= 34) {  // V (uniform per block, no barrier below this point)
        int kvh = slot - 34;
        Vt[((size_t)kvh * D_ + d) * T + t] = f2bf(x);
        return;
    }

    float ss = x * x;
#pragma unroll
    for (int off = 1; off < 32; off <<= 1) ss += __shfl_xor(ss, off, 32);
    if ((d & 31) == 0) red[d >> 5] = ss;
    __syncthreads();
    float tot = red[0] + red[1] + red[2] + red[3];
    float rms = rsqrtf(tot / 128.0f + 1e-5f);
    const float* w = (slot < 32) ? qw : kw;
    sh[d] = x * rms * w[d];
    __syncthreads();

    if (d < 64) {  // neox RoPE pair (d, d+64)
        float pos = (float)positions[t];
        float inv = __powf(10000.0f, -(float)(2 * d) / 128.0f);
        float fr = pos * inv;
        float c, s;
        __sincosf(fr, &s, &c);
        float x1 = sh[d], x2 = sh[d + 64];
        unsigned short b1 = f2bf(x1 * c - x2 * s);
        unsigned short b2 = f2bf(x2 * c + x1 * s);
        unsigned short* dst;
        if (slot < 32) dst = Q + ((size_t)slot * T + t) * D_;
        else           dst = Km + ((size_t)(slot - 32) * T + t) * D_;
        dst[d] = b1;
        dst[d + 64] = b2;
    }
}

// ---------------------------------------------------------------------------
// Flash attention (causal, GQA). grid = (T/128, H), 256 threads = 8 waves.
// Per-wave 16-row q tile; kv tiles of 32. Output written as bf16 (A-matrix
// of the O-projection GEMM).
// ---------------------------------------------------------------------------
__global__ __launch_bounds__(256) void attn_kernel(
    const unsigned short* __restrict__ Q, const unsigned short* __restrict__ Km,
    const unsigned short* __restrict__ Vt, unsigned short* __restrict__ Out,
    int T) {
    __shared__ unsigned short pbuf[8][16 * 32];  // per-wave private P tile

    const int tid = threadIdx.x, lane = tid & 31, wid = tid >> 5;
    const int half = lane >> 4, l16 = lane & 15;
    const int h = blockIdx.y;
    const int kvh = h >> 4;  // H/HKV = 16
    const int qb = blockIdx.x * 128 + wid * 16;

    const unsigned short* Qh = Q + (size_t)h * T * D_;
    const unsigned short* Kh = Km + (size_t)kvh * T * D_;
    const unsigned short* Vh = Vt + (size_t)kvh * D_ * T;

    v16bf qfr[4];
    {
        const unsigned short* base = Qh + (size_t)(qb + l16) * D_;
#pragma unroll
        for (int c = 0; c < 4; ++c) {
            v8u r;
#pragma unroll
            for (int j = 0; j < 8; ++j) {
                int k = c * 32 + 16 * half + 2 * j;
                r[j] = *(const unsigned int*)(base + k);
            }
            qfr[c] = __builtin_bit_cast(v16bf, r);
        }
    }

    v8f oacc[8];
#pragma unroll
    for (int i = 0; i < 8; ++i) oacc[i] = v8f_zero();
    float mrun[8], lrun[8];
#pragma unroll
    for (int r = 0; r < 8; ++r) { mrun[r] = -1e30f; lrun[r] = 0.0f; }

    const float scale = 0.08838834764831845f;  // 128^-0.5
    const int kend = qb + 16;

    for (int kb = 0; kb < kend; kb += 32) {
        v8f s[2];
#pragma unroll
        for (int nf = 0; nf < 2; ++nf) {
            const unsigned short* kbase = Kh + (size_t)(kb + nf * 16 + l16) * D_;
            v8f acc = v8f_zero();
#pragma unroll
            for (int c = 0; c < 4; ++c)
                acc = wmma_bf16(qfr[c], load_b16(kbase + c * 32 + 16 * half), acc);
            s[nf] = acc;
        }
#pragma unroll
        for (int nf = 0; nf < 2; ++nf) {
            int kv = kb + nf * 16 + l16;
#pragma unroll
            for (int r = 0; r < 8; ++r) {
                int q = qb + r + 8 * half;
                float v = s[nf][r] * scale;
                s[nf][r] = (kv <= q) ? v : -1e30f;
            }
        }
        float corr[8];
#pragma unroll
        for (int r = 0; r < 8; ++r) {
            float mx = fmaxf(s[0][r], s[1][r]);
#pragma unroll
            for (int off = 1; off < 16; off <<= 1)
                mx = fmaxf(mx, __shfl_xor(mx, off, 32));
            float mnew = fmaxf(mrun[r], mx);
            float cr = __expf(mrun[r] - mnew);
            float p0 = __expf(s[0][r] - mnew);
            float p1 = __expf(s[1][r] - mnew);
            s[0][r] = p0; s[1][r] = p1;
            float rs = p0 + p1;
#pragma unroll
            for (int off = 1; off < 16; off <<= 1)
                rs += __shfl_xor(rs, off, 32);
            lrun[r] = lrun[r] * cr + rs;
            mrun[r] = mnew;
            corr[r] = cr;
        }
#pragma unroll
        for (int i = 0; i < 8; ++i)
#pragma unroll
            for (int r = 0; r < 8; ++r) oacc[i][r] *= corr[r];

        unsigned short* pw = pbuf[wid];
#pragma unroll
        for (int nf = 0; nf < 2; ++nf)
#pragma unroll
            for (int r = 0; r < 8; ++r)
                pw[(r + 8 * half) * 32 + nf * 16 + l16] = f2bf(s[nf][r]);
        v16bf pa;
        {
            v8u rr;
#pragma unroll
            for (int j = 0; j < 8; ++j) {
                int k = 16 * half + 2 * j;
                rr[j] = *(const unsigned int*)(pw + l16 * 32 + k);
            }
            pa = __builtin_bit_cast(v16bf, rr);
        }
#pragma unroll
        for (int nf = 0; nf < 8; ++nf) {
            const unsigned short* vb =
                Vh + (size_t)(nf * 16 + l16) * T + kb + 16 * half;
            oacc[nf] = wmma_bf16(pa, load_b16(vb), oacc[nf]);
        }
    }

#pragma unroll
    for (int nf = 0; nf < 8; ++nf) {
        int col = h * D_ + nf * 16 + l16;
#pragma unroll
        for (int r = 0; r < 8; ++r) {
            int row = qb + r + 8 * half;
            Out[(size_t)row * (H_ * D_) + col] = f2bf(oacc[nf][r] / lrun[r]);
        }
    }
}

// ---------------------------------------------------------------------------
extern "C" void kernel_launch(void* const* d_in, const int* in_sizes, int n_in,
                              void* d_out, int out_size, void* d_ws,
                              size_t ws_size, hipStream_t stream) {
    const float* hidden = (const float*)d_in[0];
    const int* positions = (const int*)d_in[1];
    const float* qkv_w = (const float*)d_in[2];
    const float* qkv_b = (const float*)d_in[3];
    const float* qnw = (const float*)d_in[4];
    const float* knw = (const float*)d_in[5];
    const float* o_w = (const float*)d_in[6];
    float* out = (float*)d_out;

    const int T = in_sizes[0] / HID_;               // 2048
    const int QKV_N = H_ * D_ + 2 * HKV_ * D_;      // 4608

    char* ws = (char*)d_ws;
    size_t off = 0;
    float* qkv_f32 = (float*)(ws + off);  off += (size_t)T * QKV_N * 4;
    unsigned short* hid_bf = (unsigned short*)(ws + off); off += (size_t)T * HID_ * 2;
    unsigned short* qkv_wt = (unsigned short*)(ws + off); off += (size_t)QKV_N * HID_ * 2;
    unsigned short* o_wt   = (unsigned short*)(ws + off); off += (size_t)HID_ * (H_ * D_) * 2;
    unsigned short* Qb  = (unsigned short*)(ws + off); off += (size_t)H_ * T * D_ * 2;
    unsigned short* Kb  = (unsigned short*)(ws + off); off += (size_t)HKV_ * T * D_ * 2;
    unsigned short* Vtb = (unsigned short*)(ws + off); off += (size_t)HKV_ * D_ * T * 2;
    unsigned short* attn_bf = (unsigned short*)(ws + off);

    // 0. precision/layout prep: bf16 activations, transposed bf16 weights
    convert_bf16_kernel<<<4096, 256, 0, stream>>>(hidden, hid_bf,
                                                  (long)T * HID_);
    transpose_bf16_kernel<<<dim3(QKV_N / 32, HID_ / 32), 256, 0, stream>>>(
        qkv_w, qkv_wt, HID_, QKV_N);
    transpose_bf16_kernel<<<dim3(HID_ / 32, (H_ * D_) / 32), 256, 0, stream>>>(
        o_w, o_wt, H_ * D_, HID_);
    // 1. QKV projection (bias)
    gemm_bf16t_kernel<true><<<dim3(QKV_N / 256, T / 128), 256, 0, stream>>>(
        hid_bf, qkv_wt, qkv_b, qkv_f32, T, QKV_N, HID_);
    // 2. RMSNorm + RoPE + repack
    normrope_kernel<<<dim3(T, H_ + 2 * HKV_), 128, 0, stream>>>(
        qkv_f32, positions, qnw, knw, Qb, Kb, Vtb, T);
    // 3. causal GQA flash attention (bf16 out)
    attn_kernel<<<dim3(T / 128, H_), 256, 0, stream>>>(Qb, Kb, Vtb, attn_bf, T);
    // 4. output projection
    gemm_bf16t_kernel<false><<<dim3(HID_ / 256, T / 128), 256, 0, stream>>>(
        attn_bf, o_wt, nullptr, out, T, HID_, H_ * D_);
}
